// dense_res_bit_baens_61942018343228
// MI455X (gfx1250) — matrix operand order repaired and checked by reading, not compiled, not run on hardware
//
#include <hip/hip_runtime.h>

#define NEXP  8
#define D1K   1024
#define D2K   1024
#define BATCH 4096
#define UELEMS (NEXP * D1K * D2K)
#define XELEMS ((size_t)BATCH * NEXP * D1K)

typedef __attribute__((ext_vector_type(16))) __bf16          v16bf;
typedef __attribute__((ext_vector_type(8)))  __bf16          v8bf;
typedef __attribute__((ext_vector_type(2)))  __bf16          v2bf;
typedef __attribute__((ext_vector_type(8)))  float           v8f;
typedef __attribute__((ext_vector_type(4)))  float           v4f;
typedef __attribute__((ext_vector_type(8)))  unsigned short  v8us;

__device__ __forceinline__ unsigned short f2bf(float f) {
  unsigned int u = __float_as_uint(f);
  unsigned int r = u + 0x7FFFu + ((u >> 16) & 1u);
  return (unsigned short)(r >> 16);
}
__device__ __forceinline__ float bf2f(unsigned short h) {
  return __uint_as_float(((unsigned int)h) << 16);
}

__device__ __forceinline__ unsigned int pk_bf16(float a, float b) {
#if defined(__HIP_DEVICE_COMPILE__) && __has_builtin(__builtin_amdgcn_cvt_pk_bf16_f32)
  v2bf r = __builtin_amdgcn_cvt_pk_bf16_f32(a, b);
  unsigned int u;
  __builtin_memcpy(&u, &r, 4);
  return u;
#else
  return (unsigned)f2bf(a) | ((unsigned)f2bf(b) << 16);
#endif
}

__device__ __forceinline__ void split_pair(float a, float b,
                                           unsigned int& hp, unsigned int& lp) {
  hp = pk_bf16(a, b);
  float ha = __uint_as_float(hp << 16);
  float hb = __uint_as_float(hp & 0xFFFF0000u);
  lp = pk_bf16(a - ha, b - hb);
}

// ---- async global->LDS copy: real builtin on device, stub on host ---------
typedef int v4i_raw __attribute__((vector_size(16)));
typedef __attribute__((address_space(1))) v4i_raw* g128_ptr;
typedef __attribute__((address_space(3))) v4i_raw* l128_ptr;

__device__ __forceinline__ g128_ptr as_global(const void* p) {
  return (g128_ptr)(unsigned long long)(uintptr_t)p;
}
__device__ __forceinline__ l128_ptr as_lds(void* p) {
  return (l128_ptr)(unsigned int)(uintptr_t)p;
}
__device__ __forceinline__ void async_copy_b128(const void* g, void* l) {
#if defined(__HIP_DEVICE_COMPILE__) && __has_builtin(__builtin_amdgcn_global_load_async_to_lds_b128)
  __builtin_amdgcn_global_load_async_to_lds_b128(as_global(g), as_lds(l), 0, 0);
#else
  // fallback: synchronous copy through registers
  *(v4i_raw*)l = *(const v4i_raw*)g;
#endif
}
__device__ __forceinline__ void wait_async0() {
#if defined(__HIP_DEVICE_COMPILE__) && __has_builtin(__builtin_amdgcn_global_load_async_to_lds_b128)
#if __has_builtin(__builtin_amdgcn_s_wait_asynccnt)
  __builtin_amdgcn_s_wait_asynccnt(0);
#else
  asm volatile("s_wait_asynccnt 0x0" ::: "memory");
#endif
#endif
}

// ---------------------------------------------------------------------------
// Pass 1: per-block min/max of U.
// ---------------------------------------------------------------------------
__global__ __launch_bounds__(256) void minmax_partial(const float* __restrict__ U,
                                                      float* __restrict__ pmax,
                                                      float* __restrict__ pmin) {
  __shared__ float smax[256];
  __shared__ float smin[256];
  const int tid = threadIdx.x;
  const size_t base4 = (size_t)blockIdx.x * 1024;
  const v4f* U4 = (const v4f*)U;
  float mx = -3.402823466e38f, mn = 3.402823466e38f;
#pragma unroll
  for (int i = 0; i < 4; ++i) {
    v4f v = U4[base4 + tid + i * 256];
    mx = fmaxf(mx, fmaxf(fmaxf(v[0], v[1]), fmaxf(v[2], v[3])));
    mn = fminf(mn, fminf(fminf(v[0], v[1]), fminf(v[2], v[3])));
  }
  smax[tid] = mx; smin[tid] = mn;
  __syncthreads();
  for (int s = 128; s > 0; s >>= 1) {
    if (tid < s) {
      smax[tid] = fmaxf(smax[tid], smax[tid + s]);
      smin[tid] = fminf(smin[tid], smin[tid + s]);
    }
    __syncthreads();
  }
  if (tid == 0) { pmax[blockIdx.x] = smax[0]; pmin[blockIdx.x] = smin[0]; }
}

// ---------------------------------------------------------------------------
// Pass 2: final min/max reduce.
// ---------------------------------------------------------------------------
__global__ __launch_bounds__(256) void minmax_final(const float* __restrict__ pmax,
                                                    const float* __restrict__ pmin,
                                                    float* __restrict__ scal) {
  __shared__ float smax[256];
  __shared__ float smin[256];
  const int tid = threadIdx.x;
  float mx = -3.402823466e38f, mn = 3.402823466e38f;
  for (int i = tid; i < 2048; i += 256) {
    mx = fmaxf(mx, pmax[i]);
    mn = fminf(mn, pmin[i]);
  }
  smax[tid] = mx; smin[tid] = mn;
  __syncthreads();
  for (int s = 128; s > 0; s >>= 1) {
    if (tid < s) {
      smax[tid] = fmaxf(smax[tid], smax[tid + s]);
      smin[tid] = fminf(smin[tid], smin[tid + s]);
    }
    __syncthreads();
  }
  if (tid == 0) { scal[0] = smax[0]; scal[1] = smin[0]; }
}

// ---------------------------------------------------------------------------
// Pass 3: residual quantization -> bf16 hi/lo planes, TRANSPOSED (whiT[n][l][k]).
// ---------------------------------------------------------------------------
__global__ __launch_bounds__(256) void quantize_split_t(const float* __restrict__ U,
                                                        const float* __restrict__ scal,
                                                        unsigned short* __restrict__ whiT,
                                                        unsigned short* __restrict__ wloT) {
  __shared__ __attribute__((aligned(16))) unsigned short sH[64][68];
  __shared__ __attribute__((aligned(16))) unsigned short sL[64][68];
  const int tid = threadIdx.x;
  const int n  = blockIdx.z;
  const int k0 = blockIdx.y * 64;
  const int l0 = blockIdx.x * 64;

  const float beta = scal[0];
  const float alpha = scal[1];
  const float s0 = (beta - alpha) * (1.0f / 3.0f);
  const float s1 = s0 * (1.0f / 5.0f);
  const float s2 = s1 * (1.0f / 17.0f);
  const float r0 = 1.0f / s0, r1 = 1.0f / s1, r2 = 1.0f / s2;

  const float* Un = U + (size_t)n * D1K * D2K;

#pragma unroll
  for (int i = 0; i < 4; ++i) {
    int lin = tid + i * 256;
    int r   = lin >> 4;
    int c4  = lin & 15;
    v4f u = *(const v4f*)(Un + (size_t)(k0 + r) * D2K + l0 + c4 * 4);
    float w[4];
#pragma unroll
    for (int j = 0; j < 4; ++j) {
      float uv = u[j];
      float t = s0 * floorf(uv * r0);
      t += s1 * floorf((uv - t) * r1);
      t += s2 * floorf((uv - t) * r2);
      w[j] = t;
    }
    unsigned int h01, l01, h23, l23;
    split_pair(w[0], w[1], h01, l01);
    split_pair(w[2], w[3], h23, l23);
    *(uint2*)&sH[r][c4 * 4] = make_uint2(h01, h23);
    *(uint2*)&sL[r][c4 * 4] = make_uint2(l01, l23);
  }
  __syncthreads();

#pragma unroll
  for (int i = 0; i < 2; ++i) {
    int lin = tid + i * 256;
    int l   = lin >> 3;
    int c8  = lin & 7;
    unsigned short hv[8], lv[8];
#pragma unroll
    for (int j = 0; j < 8; ++j) {
      hv[j] = sH[c8 * 8 + j][l];
      lv[j] = sL[c8 * 8 + j][l];
    }
    const size_t off = (size_t)n * D1K * D2K + (size_t)(l0 + l) * D1K + k0 + c8 * 8;
    *(uint4*)(whiT + off) = make_uint4(
        (unsigned)hv[0] | ((unsigned)hv[1] << 16), (unsigned)hv[2] | ((unsigned)hv[3] << 16),
        (unsigned)hv[4] | ((unsigned)hv[5] << 16), (unsigned)hv[6] | ((unsigned)hv[7] << 16));
    *(uint4*)(wloT + off) = make_uint4(
        (unsigned)lv[0] | ((unsigned)lv[1] << 16), (unsigned)lv[2] | ((unsigned)lv[3] << 16),
        (unsigned)lv[4] | ((unsigned)lv[5] << 16), (unsigned)lv[6] | ((unsigned)lv[7] << 16));
  }
}

// ---------------------------------------------------------------------------
// Pass 3b: split x into bf16 hi/lo planes (same layout as x).
// ---------------------------------------------------------------------------
__global__ __launch_bounds__(256) void split_x(const float* __restrict__ x,
                                               unsigned short* __restrict__ xhi,
                                               unsigned short* __restrict__ xlo) {
  const size_t i0 = ((size_t)blockIdx.x * 256 + threadIdx.x) * 8;
  v4f a = *(const v4f*)(x + i0);
  v4f b = *(const v4f*)(x + i0 + 4);
  unsigned int h[4], l[4];
  split_pair(a[0], a[1], h[0], l[0]);
  split_pair(a[2], a[3], h[1], l[1]);
  split_pair(b[0], b[1], h[2], l[2]);
  split_pair(b[2], b[3], h[3], l[3]);
  *(uint4*)(xhi + i0) = make_uint4(h[0], h[1], h[2], h[3]);
  *(uint4*)(xlo + i0) = make_uint4(l[0], l[1], l[2], l[3]);
}

// ---------------------------------------------------------------------------
// GEMM common pieces.
// ---------------------------------------------------------------------------
#define BM 128
#define BN 128
#define BK 32
#define LDST 40
#define NKI (D1K / BK)

#define GEMM_COMPUTE_BODY(bsel)                                                        \
  {                                                                                    \
    v16bf ah[4], al[4], bh[2], bl[2];                                                  \
    _Pragma("unroll") for (int mi = 0; mi < 4; ++mi) {                                 \
      const unsigned short* pah = &sAhi[bsel][wm * 64 + mi * 16 + mrow][0];            \
      const unsigned short* pal = &sAlo[bsel][wm * 64 + mi * 16 + mrow][0];            \
      v8bf a0 = *(const v8bf*)(pah + kbA);                                             \
      v8bf a1 = *(const v8bf*)(pah + 16 + kbA);                                        \
      ah[mi] = __builtin_shufflevector(a0, a1, 0,1,2,3,4,5,6,7,8,9,10,11,12,13,14,15); \
      v8bf b0 = *(const v8bf*)(pal + kbA);                                             \
      v8bf b1 = *(const v8bf*)(pal + 16 + kbA);                                        \
      al[mi] = __builtin_shufflevector(b0, b1, 0,1,2,3,4,5,6,7,8,9,10,11,12,13,14,15); \
    }                                                                                  \
    _Pragma("unroll") for (int ni = 0; ni < 2; ++ni) {                                 \
      const unsigned short* pbh = &sBhi[bsel][wn * 32 + ni * 16 + mrow][0];            \
      const unsigned short* pbl = &sBlo[bsel][wn * 32 + ni * 16 + mrow][0];            \
      v8bf c0 = *(const v8bf*)(pbh + kbB);                                             \
      v8bf c1 = *(const v8bf*)(pbh + kbB + 16);                                        \
      bh[ni] = __builtin_shufflevector(c0, c1, 0,1,2,3,4,5,6,7,8,9,10,11,12,13,14,15); \
      v8bf d0 = *(const v8bf*)(pbl + kbB);                                             \
      v8bf d1 = *(const v8bf*)(pbl + kbB + 16);                                        \
      bl[ni] = __builtin_shufflevector(d0, d1, 0,1,2,3,4,5,6,7,8,9,10,11,12,13,14,15); \
    }                                                                                  \
    _Pragma("unroll") for (int mi = 0; mi < 4; ++mi) {                                 \
      _Pragma("unroll") for (int ni = 0; ni < 2; ++ni) {                               \
        acc[mi][ni] = __builtin_amdgcn_wmma_f32_16x16x32_bf16(                         \
            false, ah[mi], false, bh[ni], (short)0, acc[mi][ni], false, false);        \
        acc[mi][ni] = __builtin_amdgcn_wmma_f32_16x16x32_bf16(                         \
            false, ah[mi], false, bl[ni], (short)0, acc[mi][ni], false, false);        \
        acc[mi][ni] = __builtin_amdgcn_wmma_f32_16x16x32_bf16(                         \
            false, al[mi], false, bh[ni], (short)0, acc[mi][ni], false, false);        \
      }                                                                                \
    }                                                                                  \
  }

#define GEMM_STORE_C()                                                                 \
  {                                                                                    \
    const int rbump = (lane >= 16) ? 8 : 0;                                            \
    _Pragma("unroll") for (int mi = 0; mi < 4; ++mi) {                                 \
      _Pragma("unroll") for (int ni = 0; ni < 2; ++ni) {                               \
        int col = nbase + wn * 32 + ni * 16 + (lane & 15);                             \
        _Pragma("unroll") for (int r = 0; r < 8; ++r) {                                \
          int m = mbase + wm * 64 + mi * 16 + r + rbump;                               \
          out[(size_t)m * (NEXP * D2K) + (size_t)n * D2K + col] = acc[mi][ni][r];      \
        }                                                                              \
      }                                                                                \
    }                                                                                  \
  }

// ---------------------------------------------------------------------------
// Pass 4 (primary): fully-async GEMM over pre-split bf16 planes.
// K-loop: issue 8 async b128 per thread into other buffer, WMMA on current,
// s_wait_asynccnt 0, barrier. No conversions, no staging registers.
// ---------------------------------------------------------------------------
__global__ __launch_bounds__(256) void gemm_bf16x3_async(
    const unsigned short* __restrict__ xhi, const unsigned short* __restrict__ xlo,
    const unsigned short* __restrict__ whiT, const unsigned short* __restrict__ wloT,
    float* __restrict__ out) {
  __shared__ __attribute__((aligned(16))) unsigned short sAhi[2][BM][LDST];
  __shared__ __attribute__((aligned(16))) unsigned short sAlo[2][BM][LDST];
  __shared__ __attribute__((aligned(16))) unsigned short sBhi[2][BN][LDST];
  __shared__ __attribute__((aligned(16))) unsigned short sBlo[2][BN][LDST];

  const int tid  = threadIdx.x;
  const int lane = tid & 31;
  const int wave = tid >> 5;
  const int wm = wave & 1;
  const int wn = wave >> 1;

  const int n     = blockIdx.z;
  const int mbase = blockIdx.x * BM;
  const int nbase = blockIdx.y * BN;

  const unsigned short* XH = xhi + (size_t)n * D1K;
  const unsigned short* XL = xlo + (size_t)n * D1K;
  const unsigned short* WH = whiT + (size_t)n * D1K * D2K;
  const unsigned short* WL = wloT + (size_t)n * D1K * D2K;

  v8f acc[4][2];
  const v8f vzero = {0.f, 0.f, 0.f, 0.f, 0.f, 0.f, 0.f, 0.f};
#pragma unroll
  for (int mi = 0; mi < 4; ++mi)
#pragma unroll
    for (int ni = 0; ni < 2; ++ni) acc[mi][ni] = vzero;

  // per-thread chunk mapping: 512 b128 chunks per plane -> 2 per thread
  const int r0c = (tid + 0)   >> 2, c0c = (tid + 0)   & 3;
  const int r1c = (tid + 256) >> 2, c1c = (tid + 256) & 3;

  auto load_tiles = [&](int b, int k0) {
    const size_t ga0 = (size_t)(mbase + r0c) * (NEXP * D1K) + k0 + c0c * 8;
    const size_t ga1 = (size_t)(mbase + r1c) * (NEXP * D1K) + k0 + c1c * 8;
    async_copy_b128(XH + ga0, &sAhi[b][r0c][c0c * 8]);
    async_copy_b128(XH + ga1, &sAhi[b][r1c][c1c * 8]);
    async_copy_b128(XL + ga0, &sAlo[b][r0c][c0c * 8]);
    async_copy_b128(XL + ga1, &sAlo[b][r1c][c1c * 8]);
    const size_t gb0 = (size_t)(nbase + r0c) * D1K + k0 + c0c * 8;
    const size_t gb1 = (size_t)(nbase + r1c) * D1K + k0 + c1c * 8;
    async_copy_b128(WH + gb0, &sBhi[b][r0c][c0c * 8]);
    async_copy_b128(WH + gb1, &sBhi[b][r1c][c1c * 8]);
    async_copy_b128(WL + gb0, &sBlo[b][r0c][c0c * 8]);
    async_copy_b128(WL + gb1, &sBlo[b][r1c][c1c * 8]);
  };

  const int mrow = lane & 15;
  const int kbA  = (lane < 16) ? 0 : 8;
  const int kbB  = (lane < 16) ? 0 : 8;

  load_tiles(0, 0);
  wait_async0();
  __syncthreads();

  for (int kk = 0; kk < NKI; ++kk) {
    const int cur = kk & 1;
    if (kk + 1 < NKI) load_tiles(1 - cur, (kk + 1) * BK);
    GEMM_COMPUTE_BODY(cur)
    wait_async0();
    __syncthreads();
  }

  GEMM_STORE_C()
}

// ---------------------------------------------------------------------------
// Pass 4 (fallback, small workspace): convert x inside the GEMM.
// ---------------------------------------------------------------------------
__global__ __launch_bounds__(256) void gemm_bf16x3(const float* __restrict__ x,
                                                   const unsigned short* __restrict__ whiT,
                                                   const unsigned short* __restrict__ wloT,
                                                   float* __restrict__ out) {
  __shared__ __attribute__((aligned(16))) unsigned short sAhi[2][BM][LDST];
  __shared__ __attribute__((aligned(16))) unsigned short sAlo[2][BM][LDST];
  __shared__ __attribute__((aligned(16))) unsigned short sBhi[2][BN][LDST];
  __shared__ __attribute__((aligned(16))) unsigned short sBlo[2][BN][LDST];

  const int tid  = threadIdx.x;
  const int lane = tid & 31;
  const int wave = tid >> 5;
  const int wm = wave & 1;
  const int wn = wave >> 1;

  const int n     = blockIdx.z;
  const int mbase = blockIdx.x * BM;
  const int nbase = blockIdx.y * BN;

  const float* X = x + (size_t)n * D1K;
  const unsigned short* WHT = whiT + (size_t)n * D1K * D2K;
  const unsigned short* WLT = wloT + (size_t)n * D1K * D2K;

  v8f acc[4][2];
  const v8f vzero = {0.f, 0.f, 0.f, 0.f, 0.f, 0.f, 0.f, 0.f};
#pragma unroll
  for (int mi = 0; mi < 4; ++mi)
#pragma unroll
    for (int ni = 0; ni < 2; ++ni) acc[mi][ni] = vzero;

  v4f aR[4];
  v8us bRh[2], bRl[2];

  auto loadA = [&](int k0) {
#pragma unroll
    for (int i = 0; i < 4; ++i) {
      int lin = tid + i * 256, row = lin >> 3, c4 = lin & 7;
      aR[i] = *(const v4f*)(X + (size_t)(mbase + row) * (NEXP * D1K) + k0 + c4 * 4);
    }
  };
  auto storeA = [&](int b) {
#pragma unroll
    for (int i = 0; i < 4; ++i) {
      int lin = tid + i * 256, row = lin >> 3, c4 = lin & 7;
      unsigned int h01, l01, h23, l23;
      split_pair(aR[i][0], aR[i][1], h01, l01);
      split_pair(aR[i][2], aR[i][3], h23, l23);
      *(uint2*)&sAhi[b][row][c4 * 4] = make_uint2(h01, h23);
      *(uint2*)&sAlo[b][row][c4 * 4] = make_uint2(l01, l23);
    }
  };
  auto loadB = [&](int k0) {
#pragma unroll
    for (int i = 0; i < 2; ++i) {
      int lin = tid + i * 256, row = lin >> 2, c8 = lin & 3;
      const size_t goff = (size_t)(nbase + row) * D1K + k0 + c8 * 8;
      bRh[i] = *(const v8us*)(WHT + goff);
      bRl[i] = *(const v8us*)(WLT + goff);
    }
  };
  auto storeB = [&](int b) {
#pragma unroll
    for (int i = 0; i < 2; ++i) {
      int lin = tid + i * 256, row = lin >> 2, c8 = lin & 3;
      *(v8us*)&sBhi[b][row][c8 * 8] = bRh[i];
      *(v8us*)&sBlo[b][row][c8 * 8] = bRl[i];
    }
  };

  const int mrow = lane & 15;
  const int kbA  = (lane < 16) ? 0 : 8;
  const int kbB  = (lane < 16) ? 0 : 8;

  loadA(0);
  loadB(0);
  storeA(0);
  storeB(0);
  __syncthreads();

  for (int kk = 0; kk < NKI; ++kk) {
    const int cur = kk & 1;
    if (kk + 1 < NKI) { loadA((kk + 1) * BK); loadB((kk + 1) * BK); }
    GEMM_COMPUTE_BODY(cur)
    if (kk + 1 < NKI) { storeA(1 - cur); storeB(1 - cur); }
    __syncthreads();
  }

  GEMM_STORE_C()
}

// ---------------------------------------------------------------------------
// Launcher. Workspace layout:
//   [0..8) beta/alpha | [256..] pmax(2048) | [+8192] pmin(2048)
//   [65536..) whiT 16MB | wloT 16MB | xhi 64MB | xlo 64MB
// Falls back to the convert-in-GEMM kernel if ws_size can't hold the x planes.
// ---------------------------------------------------------------------------
extern "C" void kernel_launch(void* const* d_in, const int* in_sizes, int n_in,
                              void* d_out, int out_size, void* d_ws, size_t ws_size,
                              hipStream_t stream) {
  const float* x = (const float*)d_in[0];
  const float* U = (const float*)d_in[1];
  float* out = (float*)d_out;

  char* ws = (char*)d_ws;
  float* scal = (float*)ws;
  float* pmax = (float*)(ws + 256);
  float* pmin = (float*)(ws + 256 + 2048 * 4);
  unsigned short* whiT = (unsigned short*)(ws + 65536);
  unsigned short* wloT = whiT + (size_t)UELEMS;
  unsigned short* xhi  = wloT + (size_t)UELEMS;
  unsigned short* xlo  = xhi + XELEMS;

  const size_t need_big = 65536 + 2 * (size_t)UELEMS * 2 + 2 * XELEMS * 2;

  minmax_partial<<<2048, 256, 0, stream>>>(U, pmax, pmin);
  minmax_final<<<1, 256, 0, stream>>>(pmax, pmin, scal);

  dim3 qgrid(D2K / 64, D1K / 64, NEXP);
  quantize_split_t<<<qgrid, 256, 0, stream>>>(U, scal, whiT, wloT);

  dim3 grid(BATCH / BM, D2K / BN, NEXP);
  if (ws_size >= need_big) {
    split_x<<<(int)(XELEMS / (256 * 8)), 256, 0, stream>>>(x, xhi, xlo);
    gemm_bf16x3_async<<<grid, 256, 0, stream>>>(xhi, xlo, whiT, wloT, out);
  } else {
    gemm_bf16x3<<<grid, 256, 0, stream>>>(x, whiT, wloT, out);
  }
}
